// ContinuumMemoryCell_18769007084097
// MI455X (gfx1250) — matrix operand order repaired and checked by reading, not compiled, not run on hardware
//
#include <hip/hip_runtime.h>
#include <hip/hip_bf16.h>
#include <math.h>

typedef __bf16 bf16_t;
typedef __attribute__((ext_vector_type(16))) __bf16 v16bf;
typedef __attribute__((ext_vector_type(8)))  __bf16 v8bf;
typedef __attribute__((ext_vector_type(8)))  float  v8f;

static constexpr int BB = 16384;  // batch
static constexpr int DD = 1024;   // input dim
static constexpr int HH = 1024;   // hidden dim (== DD)

// ---------------------------------------------------------------------------
// small helpers
// ---------------------------------------------------------------------------
__global__ void zero_scalars_kernel(float* p) {
    p[0] = 0.0f;
    p[1] = 0.0f;
}

__global__ void cvt_f32_to_bf16_kernel(const float* __restrict__ s,
                                       bf16_t* __restrict__ d, int n) {
    int i = blockIdx.x * blockDim.x + threadIdx.x;
    if (i < n) d[i] = (bf16_t)s[i];
}

// dst[C,R] = transpose(src[R,C]), fp32 -> bf16, 32x32 LDS tiles
__global__ __launch_bounds__(256) void transpose_to_bf16_kernel(
    const float* __restrict__ s, bf16_t* __restrict__ d, int R, int C) {
    __shared__ float tile[32][33];
    int c0 = blockIdx.x * 32, r0 = blockIdx.y * 32;
    int tx = threadIdx.x, ty = threadIdx.y;
#pragma unroll
    for (int i = 0; i < 4; i++)
        tile[ty + 8 * i][tx] = s[(size_t)(r0 + ty + 8 * i) * C + (c0 + tx)];
    __syncthreads();
#pragma unroll
    for (int i = 0; i < 4; i++)
        d[(size_t)(c0 + ty + 8 * i) * R + (r0 + tx)] =
            (bf16_t)tile[tx][ty + 8 * i];
}

// ---------------------------------------------------------------------------
// GEMM1: v = x @ V^T, y_pred = x @ M, err = v - y_pred.
// One wave = one 32x32 super-tile of BOTH outputs (2x2 register tiling of
// 16x16 wmma tiles; A fragments shared across N, B fragments across M, and
// both GEMMs share the same A).  12 b128 loads : 8 wmma per K-step.
// ---------------------------------------------------------------------------
__global__ __launch_bounds__(256) void gemm1_kernel(
    const bf16_t* __restrict__ xb,    // [BB,DD] bf16 (rows contiguous)
    const bf16_t* __restrict__ Vb,    // [HH,DD] bf16 (row n = col n of V^T)
    const bf16_t* __restrict__ MT,    // [HH,DD] bf16 (row n = col n of M)
    float* __restrict__ vout,         // [BB,HH] fp32 (staged v, in d_out)
    bf16_t* __restrict__ err,         // [BB,HH] bf16
    bf16_t* __restrict__ errT) {      // [HH,BB] bf16
    const int lane = threadIdx.x & 31;
    const int wave = threadIdx.x >> 5;
    const int w  = blockIdx.x * 8 + wave;  // (BB/32)*(HH/32) = 16384 waves
    const int mT = w >> 5;                 // HH/32 = 32 n-supertiles
    const int nT = w & 31;
    const int m0 = mT * 32, n0 = nT * 32;
    const int hi  = lane >> 4;             // lane-half select
    const int l15 = lane & 15;

    const bf16_t* arow0 = xb + (size_t)(m0 + l15) * DD;
    const bf16_t* arow1 = arow0 + (size_t)16 * DD;
    const bf16_t* vrow0 = Vb + (size_t)(n0 + l15) * DD;
    const bf16_t* vrow1 = vrow0 + (size_t)16 * DD;
    const bf16_t* mrow0 = MT + (size_t)(n0 + l15) * DD;
    const bf16_t* mrow1 = mrow0 + (size_t)16 * DD;

    v8f accv[2][2] = {};
    v8f accy[2][2] = {};
    for (int k0 = 0; k0 < DD; k0 += 32) {
        const int ka = k0 + hi * 8;   // A: K {0..7}|{16..23} (+8 for hi half)
        const int kb = k0 + hi * 8;   // B: K {0..7}|{16..23} (+8 for hi half)
        __builtin_prefetch(arow0 + ka + 64, 0, 1);
        __builtin_prefetch(arow1 + ka + 64, 0, 1);
        union { v16bf v; v8bf h[2]; } A[2], Bv[2], Bm[2];
        A[0].h[0]  = *(const v8bf*)(arow0 + ka);
        A[0].h[1]  = *(const v8bf*)(arow0 + ka + 16);
        A[1].h[0]  = *(const v8bf*)(arow1 + ka);
        A[1].h[1]  = *(const v8bf*)(arow1 + ka + 16);
        Bv[0].h[0] = *(const v8bf*)(vrow0 + kb);
        Bv[0].h[1] = *(const v8bf*)(vrow0 + kb + 16);
        Bv[1].h[0] = *(const v8bf*)(vrow1 + kb);
        Bv[1].h[1] = *(const v8bf*)(vrow1 + kb + 16);
        Bm[0].h[0] = *(const v8bf*)(mrow0 + kb);
        Bm[0].h[1] = *(const v8bf*)(mrow0 + kb + 16);
        Bm[1].h[0] = *(const v8bf*)(mrow1 + kb);
        Bm[1].h[1] = *(const v8bf*)(mrow1 + kb + 16);
#pragma unroll
        for (int mi = 0; mi < 2; mi++) {
#pragma unroll
            for (int ni = 0; ni < 2; ni++) {
                accv[mi][ni] = __builtin_amdgcn_wmma_f32_16x16x32_bf16(
                    false, A[mi].v, false, Bv[ni].v, (short)0, accv[mi][ni],
                    false, false);
                accy[mi][ni] = __builtin_amdgcn_wmma_f32_16x16x32_bf16(
                    false, A[mi].v, false, Bm[ni].v, (short)0, accy[mi][ni],
                    false, false);
            }
        }
    }

    // C/D layout per 16x16 subtile: element r -> (M = r + hi*8, N = l15)
#pragma unroll
    for (int mi = 0; mi < 2; mi++) {
#pragma unroll
        for (int ni = 0; ni < 2; ni++) {
            const int n  = n0 + ni * 16 + l15;
            const int mb = m0 + mi * 16 + hi * 8;
            v8bf et;
#pragma unroll
            for (int r = 0; r < 8; r++) {
                float vv = accv[mi][ni][r];
                float e  = vv - accy[mi][ni][r];
                vout[(size_t)(mb + r) * HH + n] = vv;
                err[(size_t)(mb + r) * HH + n] = (bf16_t)e;
                et[r] = (bf16_t)e;
            }
            // lane holds 8 consecutive b for fixed h=n -> one 16B store
            *(v8bf*)(errT + (size_t)n * BB + mb) = et;
        }
    }
}

// ---------------------------------------------------------------------------
// Row pass: gates + mixing, one block per batch row (in-place over staged v).
// ---------------------------------------------------------------------------
__global__ __launch_bounds__(256) void row_kernel(
    const float* __restrict__ x,      // [BB,DD] fp32
    const bf16_t* __restrict__ err,   // [BB,HH] bf16
    const float* __restrict__ fgw, const float* __restrict__ fgb,
    const float* __restrict__ ugw, const float* __restrict__ ugb,
    const float* __restrict__ smw, const float* __restrict__ smb,
    float* __restrict__ out,          // d_out[0..BB*HH), holds v on entry
    float* __restrict__ scal) {       // [2]: sum(forget), sum(update)
    __shared__ float red[256];
    __shared__ float sh_mix;
    const int b = blockIdx.x;
    const int t = threadIdx.x;

    float vv[4], ee[4];
    float pf = 0.f, pu = 0.f, ps = 0.f;
#pragma unroll
    for (int i = 0; i < 4; i++) {
        int h = t + i * 256;
        float v  = out[(size_t)b * HH + h];
        float e  = (float)err[(size_t)b * HH + h];
        float xv = x[(size_t)b * DD + h];    // DD == HH
        vv[i] = v; ee[i] = e;
        pf += fgw[h] * v + fgw[HH + h] * xv;
        pu += ugw[h] * v + ugw[HH + h] * xv;
        ps += smw[h] * v;
    }
    auto block_reduce = [&](float val) -> float {
        red[t] = val;
        __syncthreads();
        for (int s = 128; s > 0; s >>= 1) {
            if (t < s) red[t] += red[t + s];
            __syncthreads();
        }
        float r = red[0];
        __syncthreads();
        return r;
    };
    float sf = block_reduce(pf);
    float su = block_reduce(pu);
    float ss = block_reduce(ps);
    if (t == 0) {
        float f = 1.f / (1.f + __expf(-(sf + fgb[0])));
        float u = 1.f / (1.f + __expf(-(su + ugb[0])));
        atomicAdd(&scal[0], f);
        atomicAdd(&scal[1], u);
        sh_mix = 1.f / (1.f + __expf(-(ss + smb[0])));
    }
    __syncthreads();
    const float mix = sh_mix;
#pragma unroll
    for (int i = 0; i < 4; i++) {
        int h = t + i * 256;
        // out = mix*v + (1-mix)*(v-e) = v - (1-mix)*e
        out[(size_t)b * HH + h] = vv[i] - (1.f - mix) * ee[i];
    }
}

// ---------------------------------------------------------------------------
// delta = err^T @ x  (K = BB), 2x2 register tiling, fused new_M epilogue.
// ---------------------------------------------------------------------------
__global__ __launch_bounds__(256) void delta_kernel(
    const bf16_t* __restrict__ errT,  // [HH,BB] bf16
    const bf16_t* __restrict__ xT,    // [DD,BB] bf16
    const float* __restrict__ M,      // [HH,DD] fp32 (original)
    const float* __restrict__ scal,   // [2]: sum(forget), sum(update)
    float* __restrict__ newM) {       // d_out + BB*HH
    const int lane = threadIdx.x & 31;
    const int wave = threadIdx.x >> 5;
    const int w  = blockIdx.x * 8 + wave;  // (HH/32)*(DD/32) = 1024 waves
    const int hT = w >> 5;                 // DD/32 = 32 d-supertiles
    const int dT = w & 31;
    const int h0 = hT * 32, d0 = dT * 32;
    const int hi  = lane >> 4;
    const int l15 = lane & 15;

    const bf16_t* arow0 = errT + (size_t)(h0 + l15) * BB;
    const bf16_t* arow1 = arow0 + (size_t)16 * BB;
    const bf16_t* brow0 = xT + (size_t)(d0 + l15) * BB;
    const bf16_t* brow1 = brow0 + (size_t)16 * BB;

    v8f acc[2][2] = {};
    for (int k0 = 0; k0 < BB; k0 += 32) {
        const int ka = k0 + hi * 8;
        const int kb = k0 + hi * 8;
        __builtin_prefetch(arow0 + ka + 64, 0, 1);
        __builtin_prefetch(brow0 + kb + 64, 0, 1);
        union { v16bf v; v8bf h[2]; } A[2], Bf[2];
        A[0].h[0]  = *(const v8bf*)(arow0 + ka);
        A[0].h[1]  = *(const v8bf*)(arow0 + ka + 16);
        A[1].h[0]  = *(const v8bf*)(arow1 + ka);
        A[1].h[1]  = *(const v8bf*)(arow1 + ka + 16);
        Bf[0].h[0] = *(const v8bf*)(brow0 + kb);
        Bf[0].h[1] = *(const v8bf*)(brow0 + kb + 16);
        Bf[1].h[0] = *(const v8bf*)(brow1 + kb);
        Bf[1].h[1] = *(const v8bf*)(brow1 + kb + 16);
#pragma unroll
        for (int mi = 0; mi < 2; mi++) {
#pragma unroll
            for (int ni = 0; ni < 2; ni++) {
                acc[mi][ni] = __builtin_amdgcn_wmma_f32_16x16x32_bf16(
                    false, A[mi].v, false, Bf[ni].v, (short)0, acc[mi][ni],
                    false, false);
            }
        }
    }

    const float cf = scal[0] * (1.0f / (float)BB);
    const float cu = scal[1] * (0.1f / ((float)BB * (float)BB));
#pragma unroll
    for (int mi = 0; mi < 2; mi++) {
#pragma unroll
        for (int ni = 0; ni < 2; ni++) {
            const int d  = d0 + ni * 16 + l15;
            const int hb = h0 + mi * 16 + hi * 8;
#pragma unroll
            for (int r = 0; r < 8; r++) {
                size_t idx = (size_t)(hb + r) * DD + d;
                newM[idx] = cf * M[idx] + cu * acc[mi][ni][r];
            }
        }
    }
}

// ---------------------------------------------------------------------------
// launch
// ---------------------------------------------------------------------------
extern "C" void kernel_launch(void* const* d_in, const int* in_sizes, int n_in,
                              void* d_out, int out_size, void* d_ws,
                              size_t ws_size, hipStream_t stream) {
    const float* x   = (const float*)d_in[0];  // [BB,DD]
    const float* Vw  = (const float*)d_in[1];  // [HH,DD]
    const float* Mm  = (const float*)d_in[2];  // [HH,DD]
    const float* fgw = (const float*)d_in[3];
    const float* fgb = (const float*)d_in[4];
    const float* ugw = (const float*)d_in[5];
    const float* ugb = (const float*)d_in[6];
    const float* smw = (const float*)d_in[7];
    const float* smb = (const float*)d_in[8];

    float* out  = (float*)d_out;                     // [BB,HH]
    float* newM = out + (size_t)BB * HH;             // [HH,DD]

    // workspace carve-up (bytes)
    char* ws = (char*)d_ws;
    bf16_t* xb   = (bf16_t*)(ws);                                   // 32 MB
    bf16_t* xT   = (bf16_t*)(ws + (size_t)BB * DD * 2);             // 32 MB
    bf16_t* err  = (bf16_t*)(ws + (size_t)BB * DD * 4);             // 32 MB
    bf16_t* errT = (bf16_t*)(ws + (size_t)BB * DD * 6);             // 32 MB
    bf16_t* Vb   = (bf16_t*)(ws + (size_t)BB * DD * 8);             //  2 MB
    bf16_t* MT   = (bf16_t*)(ws + (size_t)BB * DD * 8 + (size_t)HH * DD * 2);
    float*  scal = (float*) (ws + (size_t)BB * DD * 8 + (size_t)HH * DD * 4);

    zero_scalars_kernel<<<1, 1, 0, stream>>>(scal);

    cvt_f32_to_bf16_kernel<<<(BB * DD) / 256, 256, 0, stream>>>(x, xb, BB * DD);
    cvt_f32_to_bf16_kernel<<<(HH * DD) / 256, 256, 0, stream>>>(Vw, Vb, HH * DD);

    transpose_to_bf16_kernel<<<dim3(DD / 32, BB / 32), dim3(32, 8), 0, stream>>>(
        x, xT, BB, DD);
    transpose_to_bf16_kernel<<<dim3(DD / 32, HH / 32), dim3(32, 8), 0, stream>>>(
        Mm, MT, HH, DD);

    // (BB/32)*(HH/32) = 16384 waves, 8 waves/block
    gemm1_kernel<<<2048, 256, 0, stream>>>(xb, Vb, MT, out, err, errT);

    row_kernel<<<BB, 256, 0, stream>>>(x, err, fgw, fgb, ugw, ugb, smw, smb,
                                       out, scal);

    // (HH/32)*(DD/32) = 1024 waves
    delta_kernel<<<128, 256, 0, stream>>>(errT, xT, Mm, scal, newM);
}
